// EvollaSequenceCompressorAttention_48198122996243
// MI455X (gfx1250) — matrix operand order repaired and checked by reading, not compiled, not run on hardware
//
#include <hip/hip_runtime.h>
#include <hip/hip_bf16.h>
#include <stdint.h>

// ---------------------------------------------------------------------------
// EvollaSequenceCompressorAttention for MI455X (gfx1250, wave32, WMMA).
// Roofline: ~77 GFLOP vs ~140MB mandatory HBM traffic -> bf16 WMMA for all
// matmuls (f32 accumulate), LN/softmax in f32. Async global->LDS staging
// (ASYNCcnt) for copy-only tiles; transposed LDS staging where layout swap
// is needed.
// ---------------------------------------------------------------------------

typedef __bf16 bf16_t;
typedef __bf16 v16bf __attribute__((ext_vector_type(16)));
typedef float  v8f   __attribute__((ext_vector_type(8)));

#define DIMX 1024
#define DIM_HEAD 64
#define HEADSN 8
#define INNERX 512
#define BATCH 16
#define N1 2048
#define N2 64
#define NKEYS 2112              // N1 + N2
#define MROWS (BATCH * NKEYS)   // 33792
#define MQ (BATCH * N2)         // 1024

static __device__ __forceinline__ v8f wmma_bf16(v16bf a, v16bf b, v8f c) {
  return __builtin_amdgcn_wmma_f32_16x16x32_bf16(false, a, false, b,
                                                 (short)0, c, false, false);
}

union Frag16 { uint4 u[2]; v16bf v; bf16_t h[16]; };
union Frag32 { uint4 u[4]; bf16_t h[32]; };

// Stage 64 contiguous bytes global -> LDS via async DMA (no VGPR round trip).
// INST_OFFSET is added to BOTH the global and LDS addresses (ISA 15.18.3).
static __device__ __forceinline__ void async_copy64(unsigned lds_addr,
                                                    const bf16_t* gaddr) {
  asm volatile(
      "global_load_async_to_lds_b128 %0, %1, off\n\t"
      "global_load_async_to_lds_b128 %0, %1, off offset:16\n\t"
      "global_load_async_to_lds_b128 %0, %1, off offset:32\n\t"
      "global_load_async_to_lds_b128 %0, %1, off offset:48"
      :: "v"(lds_addr), "v"(gaddr) : "memory");
}
static __device__ __forceinline__ void wait_async() {
  asm volatile("s_wait_asynccnt 0" ::: "memory");
}
static __device__ __forceinline__ unsigned lds_off(const void* p) {
  return (unsigned)(uintptr_t)p;  // generic ptr low 32 bits = LDS offset
}

// ---------------------------------------------------------------------------
// f32 -> bf16 conversion
// ---------------------------------------------------------------------------
__global__ void cvt_bf16_kernel(const float* __restrict__ s,
                                bf16_t* __restrict__ d, int n) {
  int i = blockIdx.x * 256 + threadIdx.x;
  if (i < n) d[i] = (bf16_t)s[i];
}

// ---------------------------------------------------------------------------
// LayerNorm of x and latents -> bf16 kv_in (33792 x 1024, concat per batch)
// latents rows additionally copied into contiguous qin (1024 x 1024).
// One 256-thread block per row.
// ---------------------------------------------------------------------------
__global__ __launch_bounds__(256)
void ln_kernel(const float* __restrict__ x, const float* __restrict__ latents,
               const float* __restrict__ gm, const float* __restrict__ bm,
               const float* __restrict__ gl, const float* __restrict__ bl,
               bf16_t* __restrict__ kv_in, bf16_t* __restrict__ qin) {
  int R = blockIdx.x;
  int b = R / NKEYS;
  int i = R - b * NKEYS;
  const float* src;
  const float *g, *bb;
  if (i < N1) { src = x + ((size_t)b * N1 + i) * DIMX; g = gm; bb = bm; }
  else { src = latents + ((size_t)b * N2 + (i - N1)) * DIMX; g = gl; bb = bl; }

  int t = threadIdx.x;
  float4 v4 = ((const float4*)src)[t];
  float vv[4] = {v4.x, v4.y, v4.z, v4.w};
  float s = vv[0] + vv[1] + vv[2] + vv[3];
  float q = vv[0]*vv[0] + vv[1]*vv[1] + vv[2]*vv[2] + vv[3]*vv[3];
#pragma unroll
  for (int off = 16; off > 0; off >>= 1) {
    s += __shfl_xor(s, off, 32);
    q += __shfl_xor(q, off, 32);
  }
  __shared__ float shs[8], shq[8];
  int warp = t >> 5, lane = t & 31;
  if (lane == 0) { shs[warp] = s; shq[warp] = q; }
  __syncthreads();
  float ts = 0.f, tq = 0.f;
#pragma unroll
  for (int w = 0; w < 8; w++) { ts += shs[w]; tq += shq[w]; }
  float mu = ts * (1.0f / DIMX);
  float var = tq * (1.0f / DIMX) - mu * mu;
  float rs = rsqrtf(var + 1e-5f);

  union { bf16_t h[4]; unsigned long long u; } o;
#pragma unroll
  for (int j = 0; j < 4; j++) {
    int idx = t * 4 + j;
    o.h[j] = (bf16_t)((vv[j] - mu) * rs * g[idx] + bb[idx]);
  }
  *(unsigned long long*)(kv_in + (size_t)R * DIMX + t * 4) = o.u;
  if (i >= N1) {
    size_t qr = (size_t)(b * N2 + (i - N1)) * DIMX + t * 4;
    *(unsigned long long*)(qin + qr) = o.u;
  }
}

// ---------------------------------------------------------------------------
// Generic bf16 WMMA GEMM: C[M,N] = alpha * A[M,K] @ B[K,N]
// Block tile 128x64, 4 waves (128 thr), K step 32; each wave computes a
// 32x64 sub-tile (2 A-frags x 4 B-frags = 8 WMMAs / k-step).
// M % 128 == 0, N % 64 == 0, K % 32 == 0. Output bf16 (Cb) or f32 (Cf).
// A-fragment layout (16x32 bf16): lane = (l&15) -> M row; element e ->
//   K = e + 16*(l>=16)  => one contiguous 16-half run along K.
// B-fragment layout (32x16 bf16): lane = (l&15) -> N col; element e ->
//   K = e + 16*(l>=16)  => one contiguous 16-half run along K (Bt is n-major).
// ---------------------------------------------------------------------------
__global__ __launch_bounds__(128)
void gemm_bf16_kernel(const bf16_t* __restrict__ A, const bf16_t* __restrict__ B,
                      bf16_t* __restrict__ Cb, float* __restrict__ Cf,
                      int M, int N, int K, float alpha) {
  __shared__ __align__(16) bf16_t As[128][40];  // [m][k], 80B padded rows
  __shared__ __align__(16) bf16_t Bt[64][40];   // [n][k] transposed

  int tid = threadIdx.x;
  int w = tid >> 5, l = tid & 31;
  int lr = l & 15, hi = l >> 4;
  int n0 = blockIdx.x * 64;
  int m0 = blockIdx.y * 128;

  v8f acc[2][4] = {};

  // per-thread staging addresses
  const bf16_t* ag = A + (size_t)(m0 + tid) * K;       // one row per thread
  unsigned lds_a = lds_off(&As[tid][0]);
  int bkk = tid >> 2, bnh = (tid & 3) * 16;            // B: 32 k-rows x 64 n

  for (int k0 = 0; k0 < K; k0 += 32) {
    // --- stage A tile 128x32: async DMA, 64B contiguous per thread ---
    async_copy64(lds_a, ag + k0);
    // --- stage B tile 32x64, transposed into Bt ---
    {
      const bf16_t* bs = B + (size_t)(k0 + bkk) * N + n0 + bnh;
      Frag16 tb;
      tb.u[0] = *(const uint4*)bs;
      tb.u[1] = *(const uint4*)(bs + 8);
      if (k0 + 32 < K) __builtin_prefetch(bs + (size_t)32 * N, 0, 3);
#pragma unroll
      for (int i = 0; i < 16; i++) Bt[bnh + i][bkk] = tb.h[i];
    }
    wait_async();
    __syncthreads();

    // --- A fragments for this wave's 32 rows ---
    Frag16 fa[2];
#pragma unroll
    for (int mi = 0; mi < 2; mi++) {
      const bf16_t* p = &As[w * 32 + mi * 16 + lr][hi * 16];
      fa[mi].u[0] = *(const uint4*)p;        // K = base .. base+7
      fa[mi].u[1] = *(const uint4*)(p + 8);  // K = base+8 .. base+15
    }
    // --- 4 N-tiles of 16, 2 M-tiles each ---
#pragma unroll
    for (int nt = 0; nt < 4; nt++) {
      Frag16 fb;
      const bf16_t* p = &Bt[nt * 16 + lr][hi * 16];
      fb.u[0] = *(const uint4*)p;
      fb.u[1] = *(const uint4*)(p + 8);
      acc[0][nt] = wmma_bf16(fa[0].v, fb.v, acc[0][nt]);
      acc[1][nt] = wmma_bf16(fa[1].v, fb.v, acc[1][nt]);
    }
    __syncthreads();
  }

  // --- store (C layout: vgpr r -> M = r + 8*hi, N col = lane&15) ---
#pragma unroll
  for (int mi = 0; mi < 2; mi++)
#pragma unroll
    for (int nt = 0; nt < 4; nt++)
#pragma unroll
      for (int r = 0; r < 8; r++) {
        int m = m0 + w * 32 + mi * 16 + r + hi * 8;
        int n = n0 + nt * 16 + lr;
        float v = alpha * acc[mi][nt][r];
        if (Cf) Cf[(size_t)m * N + n] = v;
        else    Cb[(size_t)m * N + n] = (bf16_t)v;
      }
}

// ---------------------------------------------------------------------------
// Flash attention: one workgroup per (b, h); 4 waves x 16 query rows = 64 q.
// K tile staged once per workgroup via async DMA; V tile staged transposed
// ([d][key]) so P@V B-fragments are contiguous ds_load_b128 pairs.
// Online softmax; mask<=0 -> -10000 (softmax shift-invariant vs reference).
// ---------------------------------------------------------------------------
__global__ __launch_bounds__(128)
void attn_kernel(const bf16_t* __restrict__ qbuf, const bf16_t* __restrict__ kv,
                 const int* __restrict__ mask, bf16_t* __restrict__ obuf) {
  int bh = blockIdx.x;
  int b = bh >> 3, h = bh & 7;
  int tid = threadIdx.x;
  int w = tid >> 5, l = tid & 31;
  int lr = l & 15, hi = l >> 4;

  __shared__ __align__(16) bf16_t Ks[64][72];      // [key][d]   9216B
  __shared__ __align__(16) bf16_t VsT[64][72];     // [d][key]   9216B
  __shared__ __align__(16) bf16_t Pl[4][16][72];   // per-wave P 9216B

  // Q A-fragments (2 K-chunks of 32 over d), rows = w*16 + lr
  v16bf aq[2];
  {
    const bf16_t* qrow = qbuf + (size_t)(b * N2 + w * 16 + lr) * INNERX + h * DIM_HEAD;
#pragma unroll
    for (int c = 0; c < 2; c++) {
      Frag16 t;
      const bf16_t* p = qrow + c * 32 + hi * 16;
      t.u[0] = *(const uint4*)p;
      t.u[1] = *(const uint4*)(p + 8);
      aq[c] = t.v;
    }
  }

  float rmax[8], rsum[8];
  v8f acc[4] = {};
#pragma unroll
  for (int r = 0; r < 8; r++) { rmax[r] = -3.0e38f; rsum[r] = 0.0f; }

  const bf16_t* kbase = kv + (size_t)b * NKEYS * DIMX + h * DIM_HEAD;
  const bf16_t* vbase = kbase + INNERX;
  const int* mrow = mask + b * NKEYS;

  // staging split: thread -> (key row, 32-d half)
  int skey = tid >> 1, d0 = (tid & 1) * 32;
  unsigned lds_k = lds_off(&Ks[skey][d0]);

  for (int kb0 = 0; kb0 < NKEYS; kb0 += 64) {
    __syncthreads();  // previous block's K/V reads complete before restage
    // ---- stage K (async, coalesced 64B) and V (transposed) ----
    {
      size_t krow = (size_t)(kb0 + skey) * DIMX + d0;
      async_copy64(lds_k, kbase + krow);
      Frag32 tv;
      const uint4* vr = (const uint4*)(vbase + krow);
#pragma unroll
      for (int i = 0; i < 4; i++) tv.u[i] = vr[i];
#pragma unroll
      for (int i = 0; i < 32; i++) VsT[d0 + i][skey] = tv.h[i];
    }
    wait_async();
    __syncthreads();

    // ---- S = Q @ K^T for 4 key sub-tiles of 16 (B-frags from LDS) ----
    v8f s[4];
#pragma unroll
    for (int t = 0; t < 4; t++) {
      Frag16 b0, b1;
      const bf16_t* kp = &Ks[t * 16 + lr][hi * 16];
      b0.u[0] = *(const uint4*)kp;           // d chunk 0
      b0.u[1] = *(const uint4*)(kp + 8);
      b1.u[0] = *(const uint4*)(kp + 32);    // d chunk 1
      b1.u[1] = *(const uint4*)(kp + 40);
      v8f z = {};
      v8f cc = wmma_bf16(aq[0], b0.v, z);
      cc = wmma_bf16(aq[1], b1.v, cc);
      if (mrow[kb0 + t * 16 + lr] <= 0) {
#pragma unroll
        for (int r = 0; r < 8; r++) cc[r] = -10000.0f;
      }
      s[t] = cc;
    }

    // ---- block row-max across 4 tiles and 16 lanes ----
    float bmax[8];
#pragma unroll
    for (int r = 0; r < 8; r++)
      bmax[r] = fmaxf(fmaxf(s[0][r], s[1][r]), fmaxf(s[2][r], s[3][r]));
#pragma unroll
    for (int off = 8; off > 0; off >>= 1)
#pragma unroll
      for (int r = 0; r < 8; r++)
        bmax[r] = fmaxf(bmax[r], __shfl_xor(bmax[r], off, 32));

    float corr[8];
#pragma unroll
    for (int r = 0; r < 8; r++) {
      float nm = fmaxf(rmax[r], bmax[r]);
      corr[r] = __expf(rmax[r] - nm);
      rmax[r] = nm;
    }

    // ---- P = exp(S - max), block row-sum ----
    float ps[8];
#pragma unroll
    for (int r = 0; r < 8; r++) ps[r] = 0.0f;
#pragma unroll
    for (int t = 0; t < 4; t++)
#pragma unroll
      for (int r = 0; r < 8; r++) {
        float p = __expf(s[t][r] - rmax[r]);
        s[t][r] = p;
        ps[r] += p;
      }
#pragma unroll
    for (int off = 8; off > 0; off >>= 1)
#pragma unroll
      for (int r = 0; r < 8; r++) ps[r] += __shfl_xor(ps[r], off, 32);
#pragma unroll
    for (int r = 0; r < 8; r++) rsum[r] = rsum[r] * corr[r] + ps[r];
#pragma unroll
    for (int j = 0; j < 4; j++)
#pragma unroll
      for (int r = 0; r < 8; r++) acc[j][r] *= corr[r];

    // ---- C-layout -> A-layout for P via per-wave LDS ----
#pragma unroll
    for (int t = 0; t < 4; t++)
#pragma unroll
      for (int r = 0; r < 8; r++)
        Pl[w][r + hi * 8][t * 16 + lr] = (bf16_t)s[t][r];
    asm volatile("s_wait_dscnt 0" ::: "memory");  // same-wave LDS RAW

    // ---- O += P @ V (V B-frags contiguous from VsT) ----
#pragma unroll
    for (int kc = 0; kc < 2; kc++) {
      Frag16 ap;
      const bf16_t* prow = &Pl[w][lr][kc * 32 + hi * 16];
      ap.u[0] = *(const uint4*)prow;
      ap.u[1] = *(const uint4*)(prow + 8);
#pragma unroll
      for (int j = 0; j < 4; j++) {
        Frag16 bv;  // lane = d col (j*16+lr); element e -> key = kc*32+16*hi+e
        const bf16_t* vp = &VsT[j * 16 + lr][kc * 32 + hi * 16];
        bv.u[0] = *(const uint4*)vp;
        bv.u[1] = *(const uint4*)(vp + 8);
        acc[j] = wmma_bf16(ap.v, bv.v, acc[j]);
      }
    }
  }

  // ---- epilogue: O /= rowsum; scatter to (b, n2, h*64+d) bf16 ----
#pragma unroll
  for (int j = 0; j < 4; j++)
#pragma unroll
    for (int r = 0; r < 8; r++) {
      int m = w * 16 + r + hi * 8;
      int d = j * 16 + lr;
      float o = acc[j][r] / rsum[r];
      obuf[(size_t)(b * N2 + m) * INNERX + h * DIM_HEAD + d] = (bf16_t)o;
    }
}

// ---------------------------------------------------------------------------
// Launch
// ---------------------------------------------------------------------------
extern "C" void kernel_launch(void* const* d_in, const int* in_sizes, int n_in,
                              void* d_out, int out_size, void* d_ws, size_t ws_size,
                              hipStream_t stream) {
  const float* x       = (const float*)d_in[0];
  const float* latents = (const float*)d_in[1];
  const int*   mask    = (const int*)d_in[2];
  const float* gm      = (const float*)d_in[3];
  const float* bm      = (const float*)d_in[4];
  const float* gl      = (const float*)d_in[5];
  const float* bl      = (const float*)d_in[6];
  const float* Wq      = (const float*)d_in[7];
  const float* Wkv     = (const float*)d_in[8];
  const float* Wo      = (const float*)d_in[9];
  float* out = (float*)d_out;

  // workspace carve (bf16 buffers), 256B aligned
  char* p = (char*)d_ws;
  auto carve = [&](size_t bytes) {
    char* r = p;
    p += (bytes + 255) & ~(size_t)255;
    return r;
  };
  bf16_t* kv_in  = (bf16_t*)carve((size_t)MROWS * DIMX * 2);   // 69.2 MB
  bf16_t* kvbuf  = (bf16_t*)carve((size_t)MROWS * DIMX * 2);   // 69.2 MB
  bf16_t* qin    = (bf16_t*)carve((size_t)MQ * DIMX * 2);      // 2.1 MB
  bf16_t* qbuf   = (bf16_t*)carve((size_t)MQ * INNERX * 2);    // 1.0 MB
  bf16_t* obuf   = (bf16_t*)carve((size_t)MQ * INNERX * 2);    // 1.0 MB
  bf16_t* Wq_b   = (bf16_t*)carve((size_t)DIMX * INNERX * 2);  // 1.0 MB
  bf16_t* Wkv_b  = (bf16_t*)carve((size_t)DIMX * DIMX * 2);    // 2.1 MB
  bf16_t* Wo_b   = (bf16_t*)carve((size_t)INNERX * DIMX * 2);  // 1.0 MB

  // 1) weights -> bf16
  cvt_bf16_kernel<<<(DIMX * INNERX + 255) / 256, 256, 0, stream>>>(Wq, Wq_b, DIMX * INNERX);
  cvt_bf16_kernel<<<(DIMX * DIMX + 255) / 256, 256, 0, stream>>>(Wkv, Wkv_b, DIMX * DIMX);
  cvt_bf16_kernel<<<(INNERX * DIMX + 255) / 256, 256, 0, stream>>>(Wo, Wo_b, INNERX * DIMX);

  // 2) layernorm -> kv_in (+ qin)
  ln_kernel<<<MROWS, 256, 0, stream>>>(x, latents, gm, bm, gl, bl, kv_in, qin);

  // 3) kv = kv_in @ Wkv      (M=33792, N=1024, K=1024)
  gemm_bf16_kernel<<<dim3(DIMX / 64, MROWS / 128), 128, 0, stream>>>(
      kv_in, Wkv_b, kvbuf, nullptr, MROWS, DIMX, DIMX, 1.0f);

  // 4) q = ln @ Wq * scale   (M=1024, N=512, K=1024), scale = 64^-0.5
  gemm_bf16_kernel<<<dim3(INNERX / 64, MQ / 128), 128, 0, stream>>>(
      qin, Wq_b, qbuf, nullptr, MQ, INNERX, DIMX, 0.125f);

  // 5) attention per (b, h)
  attn_kernel<<<BATCH * HEADSN, 128, 0, stream>>>(qbuf, kvbuf, mask, obuf);

  // 6) out = attn_out @ Wo   (M=1024, N=1024, K=512), f32 output
  gemm_bf16_kernel<<<dim3(DIMX / 64, MQ / 128), 128, 0, stream>>>(
      obuf, Wo_b, nullptr, out, MQ, DIMX, INNERX, 1.0f);
}
